// MyModel_87522843560397
// MI455X (gfx1250) — matrix operand order repaired and checked by reading, not compiled, not run on hardware
//
#include <hip/hip_runtime.h>
#include <hip/hip_bf16.h>

typedef __attribute__((ext_vector_type(16))) _Float16 v16h;
typedef __attribute__((ext_vector_type(8)))  _Float16 v8h;
typedef __attribute__((ext_vector_type(8)))  float    v8f;

// ---------------------------------------------------------------------------
// Model / layout constants
// ---------------------------------------------------------------------------
#define MB 4096   // batch
#define MT 75     // sequence length
#define TP 96     // padded sequence length (multiple of 32 for P@V k-steps)
#define ER 64     // padded feature row (50 -> 64) for emb / lstm-out buffers
#define CR 128    // padded concat row (100 -> 128)

// ---------------------------------------------------------------------------
// WMMA helpers (CDNA5 wave32, V_WMMA_F32_16X16X32_F16)
// ---------------------------------------------------------------------------
__device__ inline v8f wmma32(v16h a, v16h b, v8f c) {
  return __builtin_amdgcn_wmma_f32_16x16x32_f16(false, a, false, b, (short)0, c,
                                                false, false);
}

// A fragment 16x32 f16: lane_lo = M; per lane the 16 halves are K = 16*hi ..
// 16*hi+15, loaded as two contiguous 8-half chunks  ->  two 16B vector loads.
__device__ inline v16h load_a16(const _Float16* src, int sm) {
  int lane = threadIdx.x & 31;
  int m = lane & 15, hi = lane >> 4;
  const _Float16* p = src + m * sm + 16 * hi;
  v8h lo = *(const v8h*)(p);
  v8h hh = *(const v8h*)(p + 8);
  return __builtin_shufflevector(lo, hh, 0, 1, 2, 3, 4, 5, 6, 7, 8, 9, 10, 11,
                                 12, 13, 14, 15);
}

// B fragment 32x16 f16 where K is the contiguous axis in memory
// (per lane: column n = lane&15, halves K = 16*hi .. 16*hi+15) -> one 32B load.
__device__ inline v16h load_b_kcontig(const _Float16* src, int ldb) {
  int lane = threadIdx.x & 31;
  int n = lane & 15, hi = lane >> 4;
  return *(const v16h*)(src + n * ldb + 16 * hi);
}

// B fragment from a pre-packed tile: tile stored as [lane][16 halves].
__device__ inline v16h load_b_packed(const _Float16* tile) {
  return *(const v16h*)(tile + (threadIdx.x & 31) * 16);
}

// B fragment where N is the contiguous axis (strided in K); branchless.
__device__ inline v16h load_b_strided(const _Float16* src, int ldb) {
  int lane = threadIdx.x & 31;
  int n = lane & 15, hi = lane >> 4;
  v16h b;
#pragma unroll
  for (int e = 0; e < 16; ++e) b[e] = src[(16 * hi + e) * ldb + n];
  return b;
}

// f32 16x16 C/D layout: lane_lo = N, row = r + 8*hi for VGPR r.
__device__ inline void store_d_lds(float* dst, int ldd, const v8f& d) {
  int lane = threadIdx.x & 31;
  int n = lane & 15, hi = lane >> 4;
#pragma unroll
  for (int r = 0; r < 8; ++r) dst[(r + 8 * hi) * ldd + n] = d[r];
}

__device__ inline float sigf(float x) { return 1.0f / (1.0f + expf(-x)); }
__device__ inline float eluf(float x) { return x > 0.0f ? x : expf(x) - 1.0f; }

// ---------------------------------------------------------------------------
// K0: embedding gather (fp32 table -> padded f16 sequences)
// ---------------------------------------------------------------------------
__global__ void gather_kernel(const int* __restrict__ in1,
                              const int* __restrict__ in2,
                              const float* __restrict__ emb,
                              _Float16* __restrict__ e1,
                              _Float16* __restrict__ e2, int n) {
  int i = blockIdx.x * blockDim.x + threadIdx.x;  // over B*T
  if (i >= n) return;
  int b = i / MT, t = i - b * MT;
  long row = ((long)b * TP + t) * ER;
  long t1 = in1[i], t2 = in2[i];
#pragma unroll 5
  for (int e = 0; e < 50; ++e) {
    e1[row + e] = (_Float16)emb[t1 * 50 + e];
    e2[row + e] = (_Float16)emb[t2 * 50 + e];
  }
}

// ---------------------------------------------------------------------------
// Kp: pack a f32 weight [K, G] into WMMA-B-fragment order, zero padded.
// ---------------------------------------------------------------------------
__global__ void pack_b_kernel(const float* __restrict__ src,
                              _Float16* __restrict__ dst, int K, int G, int KT,
                              int NT) {
  int total = KT * NT * 512;
  for (int i = blockIdx.x * blockDim.x + threadIdx.x; i < total;
       i += gridDim.x * blockDim.x) {
    int e = i & 15;
    int lane = (i >> 4) & 31;
    int tile = i >> 9;
    int nt = tile % NT, kt = tile / NT;
    int gk = kt * 32 + (((lane >> 4) << 4) + e);
    int gn = nt * 16 + (lane & 15);
    float v = (gk < K && gn < G) ? src[(long)gk * G + gn] : 0.0f;
    dst[i] = (_Float16)v;
  }
}

// ---------------------------------------------------------------------------
// K1/K3: scaled dot-product attention, d=50 (padded 64), T=75 (padded 96).
// One wave handles one (batch, 16-row query tile).
// ---------------------------------------------------------------------------
template <int WAVES>
__global__ __launch_bounds__(WAVES * 32) void attn_kernel(
    const _Float16* __restrict__ q,   // [B, TP, ER]
    const _Float16* __restrict__ kv,  // [B, TP, ER]
    const float* __restrict__ scale_p,
    _Float16* __restrict__ out,       // [B, TP, CR]
    int col_off) {
  constexpr int QT = 5;   // ceil(75/16) query/key tiles
  constexpr int KD = 2;   // 64/32 feature k-steps
  constexpr int KT3 = 3;  // 96/32 key k-steps for P@V
  constexpr int NV = 4;   // 64/16 value column tiles

  __shared__ float    sbuf[WAVES][16][QT * 16];  // raw scores
  __shared__ _Float16 pbuf[WAVES][16][TP];       // softmaxed P (f16, padded)

  const int wave = threadIdx.x >> 5;
  const int lane = threadIdx.x & 31;
  const long flat = (long)blockIdx.x * WAVES + wave;
  const int b = (int)(flat / QT);
  const int qt = (int)(flat % QT);
  const int m0 = qt * 16;
  const float scale = scale_p[0];

  const _Float16* qb = q + ((long)b * TP + m0) * ER;
  const _Float16* kb = kv + (long)b * TP * ER;

  // S = Q @ K^T
  for (int nt = 0; nt < QT; ++nt) {
    v8f acc = {};
#pragma unroll
    for (int kx = 0; kx < KD; ++kx) {
      v16h a = load_a16(qb + kx * 32, ER);
      v16h bf = load_b_kcontig(kb + nt * 16 * ER + kx * 32, ER);
      acc = wmma32(a, bf, acc);
    }
    store_d_lds(&sbuf[wave][0][nt * 16], QT * 16, acc);
  }

  // per-row softmax over the 75 valid columns; emit f16 P, zero padding
  if (lane < 16) {
    float mx = -1e30f;
    for (int k = 0; k < MT; ++k) mx = fmaxf(mx, scale * sbuf[wave][lane][k]);
    float sum = 0.0f;
    for (int k = 0; k < MT; ++k) {
      float p = expf(scale * sbuf[wave][lane][k] - mx);
      sbuf[wave][lane][k] = p;
      sum += p;
    }
    float inv = 1.0f / sum;
    for (int k = 0; k < MT; ++k)
      pbuf[wave][lane][k] = (_Float16)(sbuf[wave][lane][k] * inv);
    for (int k = MT; k < TP; ++k) pbuf[wave][lane][k] = (_Float16)0.0f;
  }

  // O = P @ V  (padded rows/cols of V are zero)
  for (int vt = 0; vt < NV; ++vt) {
    v8f acc = {};
#pragma unroll
    for (int kt = 0; kt < KT3; ++kt) {
      v16h a = load_a16(&pbuf[wave][0][kt * 32], TP);
      v16h bf = load_b_strided(kb + (long)kt * 32 * ER + vt * 16, ER);
      acc = wmma32(a, bf, acc);
    }
    int n = lane & 15, hi = lane >> 4;
#pragma unroll
    for (int r = 0; r < 8; ++r) {
      int m = r + 8 * hi;
      out[((long)b * TP + m0 + m) * CR + col_off + vt * 16 + n] =
          (_Float16)acc[r];
    }
  }
}

// ---------------------------------------------------------------------------
// LSTM: per-direction parameters; up to 4 directions per launch (blockIdx.y).
// ---------------------------------------------------------------------------
struct LstmDir {
  const _Float16* x;   // [B, TP, XR] f16 padded
  const _Float16* Wp;  // packed W fragments
  const _Float16* Up;  // packed U fragments
  const float* bias;   // [4*UNITS]
  _Float16* out_seq;   // [B, TP, ER] (SEQ)
  float* out_last;     // [B, 160]   (!SEQ)
  int col_off;
  int reverse;
};

__device__ inline LstmDir pick_dir(const LstmDir& a, const LstmDir& b,
                                   const LstmDir& c, const LstmDir& d, int i) {
  if (i == 1) return b;
  if (i == 2) return c;
  if (i == 3) return d;
  return a;
}

constexpr int align256i(int x) { return (x + 255) & ~255; }
struct LstmOffs { int oU, oB, oZ, oC, oH, total; };
constexpr LstmOffs lstm_offs(int XR, int UNITS, int WAVES) {
  const int G = 4 * UNITS;
  const int NT = (G + 15) / 16;
  const int KX = XR / 32;
  const int KH = (UNITS + 31) / 32;
  const int HR = KH * 32;
  LstmOffs o{};
  o.oU = align256i(KX * NT * 512 * 2);                   // after Wl
  o.oB = align256i(o.oU + KH * NT * 512 * 2);            // after Ul
  o.oZ = align256i(o.oB + G * 4);                        // after bias
  o.oC = align256i(o.oZ + WAVES * 16 * NT * 16 * 4);     // after zbuf
  o.oH = align256i(o.oC + WAVES * 16 * UNITS * 4);       // after cbuf
  o.total = align256i(o.oH + WAVES * 16 * HR * 2);       // after hbuf
  return o;
}

// One wave owns 16 batch rows; weights/bias staged in LDS once per block;
// h/c live in per-wave LDS; per timestep z = x_t@W + h@U via WMMA.
template <int XR, int UNITS, int WAVES, bool SEQ>
__global__ __launch_bounds__(WAVES * 32) void lstm_kernel(LstmDir d0,
                                                          LstmDir d1,
                                                          LstmDir d2,
                                                          LstmDir d3) {
  constexpr int G = 4 * UNITS;
  constexpr int NT = (G + 15) / 16;      // gate column tiles
  constexpr int KX = XR / 32;            // k-steps for x@W (padded)
  constexpr int KH = (UNITS + 31) / 32;  // k-steps for h@U
  constexpr int HR = KH * 32;            // padded h row
  constexpr int NT16 = NT * 16;
  constexpr LstmOffs O = lstm_offs(XR, UNITS, WAVES);

  extern __shared__ char smem[];
  _Float16* Wl = (_Float16*)(smem);
  _Float16* Ul = (_Float16*)(smem + O.oU);
  float*    bl = (float*)(smem + O.oB);

  const LstmDir dd = pick_dir(d0, d1, d2, d3, blockIdx.y);

  // cooperative stage of packed weights + bias into LDS
  {
    constexpr int nW4 = KX * NT * 512 / 8;  // uint4 count
    constexpr int nU4 = KH * NT * 512 / 8;
    const uint4* ws4 = (const uint4*)dd.Wp;
    const uint4* us4 = (const uint4*)dd.Up;
    uint4* wd4 = (uint4*)Wl;
    uint4* ud4 = (uint4*)Ul;
    for (int i = threadIdx.x; i < nW4; i += WAVES * 32) wd4[i] = ws4[i];
    for (int i = threadIdx.x; i < nU4; i += WAVES * 32) ud4[i] = us4[i];
    for (int i = threadIdx.x; i < G; i += WAVES * 32) bl[i] = dd.bias[i];
  }
  __syncthreads();

  const int wave = threadIdx.x >> 5;
  const int lane = threadIdx.x & 31;
  const long row0 = ((long)blockIdx.x * WAVES + wave) * 16;

  float*    zbuf = (float*)(smem + O.oZ) + wave * 16 * NT16;
  float*    cbuf = (float*)(smem + O.oC) + wave * 16 * UNITS;
  _Float16* hbuf = (_Float16*)(smem + O.oH) + wave * 16 * HR;

  for (int i = lane; i < 16 * UNITS; i += 32) cbuf[i] = 0.0f;
  for (int i = lane; i < 16 * HR; i += 32) hbuf[i] = (_Float16)0.0f;
  // Per-wave LDS traffic below: same-wave DS ops are in-order, no barrier.

  for (int s = 0; s < MT; ++s) {
    const int t = dd.reverse ? (MT - 1 - s) : s;
    const _Float16* xt = dd.x + (row0 * TP + t) * XR;

    for (int nt = 0; nt < NT; ++nt) {
      v8f acc = {};
#pragma unroll
      for (int kx = 0; kx < KX; ++kx) {
        v16h a = load_a16(xt + kx * 32, TP * XR);
        v16h bf = load_b_packed(Wl + (kx * NT + nt) * 512);
        acc = wmma32(a, bf, acc);
      }
#pragma unroll
      for (int kh = 0; kh < KH; ++kh) {
        v16h a = load_a16(hbuf + kh * 32, HR);
        v16h bf = load_b_packed(Ul + (kh * NT + nt) * 512);
        acc = wmma32(a, bf, acc);
      }
      store_d_lds(zbuf + nt * 16, NT16, acc);
    }

    // fused gate math: Keras order i, f, c~, o; elu cell activation
    for (int i = lane; i < 16 * UNITS; i += 32) {
      int r = i / UNITS, u = i - r * UNITS;
      const float* zr = zbuf + r * NT16;
      float zi = zr[u]             + bl[u];
      float zf = zr[UNITS + u]     + bl[UNITS + u];
      float zg = zr[2 * UNITS + u] + bl[2 * UNITS + u];
      float zo = zr[3 * UNITS + u] + bl[3 * UNITS + u];
      float c = sigf(zf) * cbuf[i] + sigf(zi) * eluf(zg);
      float h = sigf(zo) * eluf(c);
      cbuf[i] = c;
      hbuf[r * HR + u] = (_Float16)h;
      if (SEQ) {
        dd.out_seq[((row0 + r) * TP + t) * ER + dd.col_off + u] = (_Float16)h;
      } else if (s == MT - 1) {
        dd.out_last[(row0 + r) * 160 + dd.col_off + u] = h;
      }
    }
  }
}

// ---------------------------------------------------------------------------
// K5: dense [B,160] @ [160,1] + bias, sigmoid
// ---------------------------------------------------------------------------
__global__ void dense_kernel(const float* __restrict__ h,
                             const float* __restrict__ Wd,
                             const float* __restrict__ bd,
                             float* __restrict__ out, int nb) {
  int i = blockIdx.x * blockDim.x + threadIdx.x;
  if (i >= nb) return;
  float s = bd[0];
#pragma unroll 8
  for (int j = 0; j < 160; ++j) s += h[(long)i * 160 + j] * Wd[j];
  out[i] = 1.0f / (1.0f + expf(-s));
}

// ---------------------------------------------------------------------------
extern "C" void kernel_launch(void* const* d_in, const int* in_sizes, int n_in,
                              void* d_out, int out_size, void* d_ws,
                              size_t ws_size, hipStream_t stream) {
  (void)in_sizes; (void)n_in; (void)out_size; (void)ws_size;

  const int*   in1 = (const int*)d_in[0];
  const int*   in2 = (const int*)d_in[1];
  const float* emb = (const float*)d_in[2];
  const float *W1f = (const float*)d_in[3],  *U1f = (const float*)d_in[4],
              *b1f = (const float*)d_in[5];
  const float *W1b = (const float*)d_in[6],  *U1b = (const float*)d_in[7],
              *b1b = (const float*)d_in[8];
  const float *W2f = (const float*)d_in[9],  *U2f = (const float*)d_in[10],
              *b2f = (const float*)d_in[11];
  const float *W2b = (const float*)d_in[12], *U2b = (const float*)d_in[13],
              *b2b = (const float*)d_in[14];
  const float *scale1 = (const float*)d_in[15], *scale2 = (const float*)d_in[16];
  const float *W3f = (const float*)d_in[17], *U3f = (const float*)d_in[18],
              *b3f = (const float*)d_in[19];
  const float *W3b = (const float*)d_in[20], *U3b = (const float*)d_in[21],
              *b3b = (const float*)d_in[22];
  const float *Wd = (const float*)d_in[23], *bd = (const float*)d_in[24];
  float* outp = (float*)d_out;

  char* ws = (char*)d_ws;
  size_t off = 0;
  auto carve = [&](size_t bytes) {
    void* p = ws + off;
    off = (off + bytes + 255) & ~(size_t)255;
    return p;
  };
  const size_t seq_bytes = (size_t)MB * TP * ER * sizeof(_Float16);  // 48MB
  const size_t cat_bytes = (size_t)MB * TP * CR * sizeof(_Float16);  // 96MB
  _Float16* e1   = (_Float16*)carve(seq_bytes);
  _Float16* e2   = (_Float16*)carve(seq_bytes);
  _Float16* out1 = (_Float16*)carve(seq_bytes);
  _Float16* out2 = (_Float16*)carve(seq_bytes);
  _Float16* cat  = (_Float16*)carve(cat_bytes);
  float*    h3   = (float*)carve((size_t)MB * 160 * sizeof(float));
  const int SWP = 2 * 7 * 512, SUP = 1 * 7 * 512;
  const int BWP = 4 * 20 * 512, BUP = 3 * 20 * 512;
  _Float16* W1fp = (_Float16*)carve(SWP * 2);
  _Float16* U1fp = (_Float16*)carve(SUP * 2);
  _Float16* W1bp = (_Float16*)carve(SWP * 2);
  _Float16* U1bp = (_Float16*)carve(SUP * 2);
  _Float16* W2fp = (_Float16*)carve(SWP * 2);
  _Float16* U2fp = (_Float16*)carve(SUP * 2);
  _Float16* W2bp = (_Float16*)carve(SWP * 2);
  _Float16* U2bp = (_Float16*)carve(SUP * 2);
  _Float16* W3fp = (_Float16*)carve(BWP * 2);
  _Float16* U3fp = (_Float16*)carve(BUP * 2);
  _Float16* W3bp = (_Float16*)carve(BWP * 2);
  _Float16* U3bp = (_Float16*)carve(BUP * 2);

  // zero the padded sequence buffers (padding must contribute 0 to WMMAs)
  hipMemsetAsync(e1, 0, seq_bytes, stream);
  hipMemsetAsync(e2, 0, seq_bytes, stream);
  hipMemsetAsync(out1, 0, seq_bytes, stream);
  hipMemsetAsync(out2, 0, seq_bytes, stream);
  hipMemsetAsync(cat, 0, cat_bytes, stream);

  // K0: gather embeddings
  {
    int n = MB * MT;
    gather_kernel<<<(n + 255) / 256, 256, 0, stream>>>(in1, in2, emb, e1, e2,
                                                       n);
  }

  // Kp: pack all weights into WMMA B-fragment order
  pack_b_kernel<<<32, 256, 0, stream>>>(W1f, W1fp, 50, 100, 2, 7);
  pack_b_kernel<<<32, 256, 0, stream>>>(U1f, U1fp, 25, 100, 1, 7);
  pack_b_kernel<<<32, 256, 0, stream>>>(W1b, W1bp, 50, 100, 2, 7);
  pack_b_kernel<<<32, 256, 0, stream>>>(U1b, U1bp, 25, 100, 1, 7);
  pack_b_kernel<<<32, 256, 0, stream>>>(W2f, W2fp, 50, 100, 2, 7);
  pack_b_kernel<<<32, 256, 0, stream>>>(U2f, U2fp, 25, 100, 1, 7);
  pack_b_kernel<<<32, 256, 0, stream>>>(W2b, W2bp, 50, 100, 2, 7);
  pack_b_kernel<<<32, 256, 0, stream>>>(U2b, U2bp, 25, 100, 1, 7);
  pack_b_kernel<<<64, 256, 0, stream>>>(W3f, W3fp, 100, 320, 4, 20);
  pack_b_kernel<<<64, 256, 0, stream>>>(U3f, U3fp, 80, 320, 3, 20);
  pack_b_kernel<<<64, 256, 0, stream>>>(W3b, W3bp, 100, 320, 4, 20);
  pack_b_kernel<<<64, 256, 0, stream>>>(U3b, U3bp, 80, 320, 3, 20);

  // K1: attn1(emb1, emb2) -> cat[:, :, 0:50]
  attn_kernel<4><<<MB * 5 / 4, 128, 0, stream>>>(e1, e2, scale1, cat, 0);

  // K2: all 4 small-LSTM directions concurrently (blockIdx.y = direction)
  {
    LstmDir s0{e1, W1fp, U1fp, b1f, out1, nullptr, 0, 0};
    LstmDir s1{e1, W1bp, U1bp, b1b, out1, nullptr, 25, 1};
    LstmDir s2{e2, W2fp, U2fp, b2f, out2, nullptr, 0, 0};
    LstmDir s3{e2, W2bp, U2bp, b2b, out2, nullptr, 25, 1};
    constexpr LstmOffs OS = lstm_offs(ER, 25, 4);
    dim3 gs(MB / (16 * 4), 4);
    lstm_kernel<ER, 25, 4, true><<<gs, 128, OS.total, stream>>>(s0, s1, s2, s3);
  }

  // K3: attn2(out1, out2) -> cat[:, :, 50:100]
  attn_kernel<4><<<MB * 5 / 4, 128, 0, stream>>>(out1, out2, scale2, cat, 50);

  // K4: both big-LSTM directions concurrently, last hidden -> h3 [B,160]
  {
    LstmDir g0{cat, W3fp, U3fp, b3f, nullptr, h3, 0, 0};
    LstmDir g1{cat, W3bp, U3bp, b3b, nullptr, h3, 80, 1};
    constexpr LstmOffs OB = lstm_offs(CR, 80, 2);
    dim3 gb(MB / (16 * 2), 2);
    lstm_kernel<CR, 80, 2, false><<<gb, 64, OB.total, stream>>>(g0, g1, g0, g1);
  }

  // K5: dense + sigmoid -> d_out [B,1]
  dense_kernel<<<MB / 256, 256, 0, stream>>>(h3, Wd, bd, outp, MB);
}
